// RecurrentChannelAttentionV3_50998441672742
// MI455X (gfx1250) — matrix operand order repaired and checked by reading, not compiled, not run on hardware
//
#include <hip/hip_runtime.h>
#include <hip/hip_bf16.h>
#include <stdint.h>

// ---------------------------------------------------------------------------
// Types
// ---------------------------------------------------------------------------
typedef __bf16 bf16_t;
typedef __bf16  v16bf __attribute__((ext_vector_type(16)));
typedef __bf16  v8bf  __attribute__((ext_vector_type(8)));
typedef float   v8f   __attribute__((ext_vector_type(8)));
typedef unsigned int u32x4 __attribute__((ext_vector_type(4)));
typedef int          i32x4 __attribute__((ext_vector_type(4)));
typedef int          i32x8 __attribute__((ext_vector_type(8)));

#if defined(__HIP_DEVICE_COMPILE__) && __has_builtin(__builtin_amdgcn_tensor_load_to_lds) && __has_builtin(__builtin_amdgcn_s_wait_tensorcnt)
#define HAS_TDM 1
#else
#define HAS_TDM 0
#endif

// Problem constants
#define TT 8192
#define CC 512
#define GG 2048   // 4*C
#define MM 5

// ---------------------------------------------------------------------------
// Helpers
// ---------------------------------------------------------------------------
__device__ __forceinline__ v16bf cat8(v8bf a, v8bf b) {
  return __builtin_shufflevector(a, b, 0,1,2,3,4,5,6,7,8,9,10,11,12,13,14,15);
}
__device__ __forceinline__ float sigm(float x) { return 1.0f / (1.0f + __expf(-x)); }

// TDM: load a k-slab Whh[:, kc*32 .. kc*32+31] (2048 rows x 32 bf16 cols) into LDS.
__device__ __forceinline__ void tdm_issue_slab(const bf16_t* whh, int kc, unsigned lds_off) {
#if HAS_TDM
  unsigned long long ga = (unsigned long long)(uintptr_t)whh + (unsigned long long)(kc * 64);
  u32x4 g0;
  g0[0] = 1u;                                   // count=1, user descriptor
  g0[1] = lds_off;                              // lds_addr (bytes)
  g0[2] = (unsigned)ga;                         // global_addr[31:0]
  g0[3] = (unsigned)((ga >> 32) & 0x01FFFFFFull) | 0x80000000u; // addr[56:32] | type=2
  i32x8 g1;
  g1[0] = (int)(1u << 16);                      // workgroup_mask=0, data_size=1 (2B)
  g1[1] = (int)(512u << 16);                    // atomic_addr=0 | tensor_dim0.lo16 (512)
  g1[2] = (int)(2048u << 16);                   // tensor_dim0.hi16=0 | tensor_dim1.lo16 (2048)
  g1[3] = (int)(32u << 16);                     // tensor_dim1.hi16=0 | tile_dim0 (32)
  g1[4] = (int)(2048u);                         // tile_dim1=2048, tile_dim2=0
  g1[5] = 512;                                  // tensor_dim0_stride lo32 (512 elems)
  g1[6] = 0;
  g1[7] = 0;
  i32x4 z4 = {0, 0, 0, 0};
#if defined(__clang_major__) && (__clang_major__ >= 23)
  i32x8 z8 = {0, 0, 0, 0, 0, 0, 0, 0};
  __builtin_amdgcn_tensor_load_to_lds(g0, g1, z4, z4, z8, 0);
#else
  __builtin_amdgcn_tensor_load_to_lds(g0, g1, z4, z4, 0);
#endif
#else
  (void)whh; (void)kc; (void)lds_off;
#endif
}

// ---------------------------------------------------------------------------
// K0: conversions
// ---------------------------------------------------------------------------
__global__ void k_cvt_bf16(const float* __restrict__ src, bf16_t* __restrict__ dst, int n) {
  for (int i = blockIdx.x * blockDim.x + threadIdx.x; i < n; i += gridDim.x * blockDim.x)
    dst[i] = (bf16_t)src[i];
}
__global__ void k_cvt_bf16_strided(const float* __restrict__ src, bf16_t* __restrict__ dst,
                                   int rows, int cols, int src_stride, int src_off) {
  int n = rows * cols;
  for (int i = blockIdx.x * blockDim.x + threadIdx.x; i < n; i += gridDim.x * blockDim.x) {
    int r = i / cols, k = i - r * cols;
    dst[i] = (bf16_t)src[(size_t)r * src_stride + src_off + k];
  }
}
__global__ void k_bias_sum(const float* __restrict__ a, const float* __restrict__ b,
                           float* __restrict__ dst, int n) {
  for (int i = blockIdx.x * blockDim.x + threadIdx.x; i < n; i += gridDim.x * blockDim.x)
    dst[i] = a[i] + b[i];
}

// ---------------------------------------------------------------------------
// K1: input projection pregates  XG[t][m][n] = x[m,t,:]@Wih^T[:,n] + bias[n]
// ---------------------------------------------------------------------------
__global__ void __launch_bounds__(256) k_input_proj(const bf16_t* __restrict__ X,
                                                    const bf16_t* __restrict__ Wih,
                                                    const float* __restrict__ bias,
                                                    bf16_t* __restrict__ XG, int B) {
  const int lane  = threadIdx.x & 31;
  const int wave  = threadIdx.x >> 5;
  const int rt    = blockIdx.y;
  const int tile  = blockIdx.x * 8 + wave;   // 0..127
  const int nbase = tile * 16;
  const int sub   = lane & 15;
  const int kh    = lane >> 4;

  const bf16_t* arow = X + (size_t)(rt * 16 + sub) * CC;
  const bf16_t* brow = Wih + (size_t)(nbase + sub) * CC;

  v8f acc = {0.f, 0.f, 0.f, 0.f, 0.f, 0.f, 0.f, 0.f};
  for (int kc = 0; kc < 16; ++kc) {
    const bf16_t* ap = arow + kc * 32 + kh * 16;
    v16bf a = cat8(*(const v8bf*)ap, *(const v8bf*)(ap + 8));
    const bf16_t* bp = brow + kc * 32 + kh * 16;
    v16bf b = cat8(*(const v8bf*)bp, *(const v8bf*)(bp + 8));
    acc = __builtin_amdgcn_wmma_f32_16x16x32_bf16(false, a, false, b, (short)0, acc, false, false);
  }
  const int n  = nbase + sub;
  const float bn = bias[n];
#pragma unroll
  for (int r = 0; r < 8; ++r) {
    int rg = rt * 16 + ((lane < 16) ? r : (r + 8));
    int m  = rg >> 13;        // rg / T
    int t  = rg & (TT - 1);
    XG[((size_t)t * B + m) * GG + n] = (bf16_t)(acc[r] + bn);
  }
}

// ---------------------------------------------------------------------------
// K1b: P[t][n] = Hq_bf[t,:] @ W1^T[:,n] + Web[n]   (N=512, K=512, f32 out)
// grid (4 colgroups, T/16 rowtiles), block 256 (8 waves)
// ---------------------------------------------------------------------------
__global__ void __launch_bounds__(256) k_gemm_p(const bf16_t* __restrict__ X,
                                                const bf16_t* __restrict__ W1,
                                                const float* __restrict__ Web,
                                                float* __restrict__ P) {
  const int lane  = threadIdx.x & 31;
  const int wave  = threadIdx.x >> 5;
  const int rt    = blockIdx.y;
  const int tile  = blockIdx.x * 8 + wave;   // 0..31
  const int nbase = tile * 16;
  const int sub   = lane & 15;
  const int kh    = lane >> 4;

  const bf16_t* arow = X + (size_t)(rt * 16 + sub) * CC;
  const bf16_t* brow = W1 + (size_t)(nbase + sub) * CC;

  v8f acc = {0.f, 0.f, 0.f, 0.f, 0.f, 0.f, 0.f, 0.f};
  for (int kc = 0; kc < 16; ++kc) {
    const bf16_t* ap = arow + kc * 32 + kh * 16;
    v16bf a = cat8(*(const v8bf*)ap, *(const v8bf*)(ap + 8));
    const bf16_t* bp = brow + kc * 32 + kh * 16;
    v16bf b = cat8(*(const v8bf*)bp, *(const v8bf*)(bp + 8));
    acc = __builtin_amdgcn_wmma_f32_16x16x32_bf16(false, a, false, b, (short)0, acc, false, false);
  }
  const int n  = nbase + sub;
  const float bn = Web[n];
#pragma unroll
  for (int r = 0; r < 8; ++r) {
    int t = rt * 16 + ((lane < 16) ? r : (r + 8));
    P[(size_t)t * CC + n] = acc[r] + bn;
  }
}

// ---------------------------------------------------------------------------
// K2: persistent recurrent LSTM. One block per LSTM, 512 threads (16 waves).
// ---------------------------------------------------------------------------
struct LstmArgs {
  const bf16_t* Whh;   // [2048][512] bf16
  const bf16_t* XG;    // [T][B][2048] bf16 pregates (incl. bias)
  float* Hout;         // [T][B][512] f32
  int B;
};

#define SLAB_BYTES 131072u            // 2048*32*2
#define HLDS_OFF   262144u
#define DYN_LDS    278528u            // 2*slab + 16KB H

__global__ void __launch_bounds__(512) k_lstm(LstmArgs a0, LstmArgs a1, LstmArgs a2) {
  LstmArgs a = (blockIdx.x == 0) ? a0 : ((blockIdx.x == 1) ? a1 : a2);
  extern __shared__ char smem[];
  bf16_t* bufs[2] = {(bf16_t*)smem, (bf16_t*)(smem + SLAB_BYTES)};
  bf16_t* H = (bf16_t*)(smem + HLDS_OFF);   // [16][512]

  const int lane = threadIdx.x & 31;
  const int w    = threadIdx.x >> 5;   // wave 0..15
  const int sub  = lane & 15;
  const int kh   = lane >> 4;
  const int B    = a.B;

  for (int i = threadIdx.x; i < 16 * CC; i += 512) H[i] = (bf16_t)0.0f;
  __syncthreads();

  float cst[2][5];
#pragma unroll
  for (int u = 0; u < 2; ++u)
#pragma unroll
    for (int r = 0; r < 5; ++r) cst[u][r] = 0.f;

#if HAS_TDM
  if (w == 0) tdm_issue_slab(a.Whh, 0, 0u);
#endif

  for (int t = 0; t < TT; ++t) {
    v8f acc[8];
#pragma unroll
    for (int s = 0; s < 8; ++s) acc[s] = (v8f){0.f, 0.f, 0.f, 0.f, 0.f, 0.f, 0.f, 0.f};

    for (int kc = 0; kc < 16; ++kc) {
#if HAS_TDM
      if (w == 0) {
        bool more;
        if (kc < 15) { tdm_issue_slab(a.Whh, kc + 1, ((kc + 1) & 1) * SLAB_BYTES); more = true; }
        else if (t + 1 < TT) { tdm_issue_slab(a.Whh, 0, 0u); more = true; }
        else more = false;
        if (more) __builtin_amdgcn_s_wait_tensorcnt(1);
        else      __builtin_amdgcn_s_wait_tensorcnt(0);
      }
#else
      {  // cooperative fallback copy
        uint4* dst = (uint4*)bufs[kc & 1];
        for (int i = threadIdx.x; i < 8192; i += 512) {
          int row = i >> 2, q = i & 3;
          dst[i] = *(const uint4*)(a.Whh + (size_t)row * CC + kc * 32 + q * 8);
        }
      }
#endif
      __syncthreads();   // slab ready

      const bf16_t* hp = H + sub * CC + kc * 32 + kh * 16;
      v16bf afrag = cat8(*(const v8bf*)hp, *(const v8bf*)(hp + 8));
      const bf16_t* sb = bufs[kc & 1];
#pragma unroll
      for (int s = 0; s < 8; ++s) {
        const int tl = w + 16 * s;
        const bf16_t* bp = sb + (size_t)(tl * 16 + sub) * 32 + kh * 16;
        v16bf bfrag = cat8(*(const v8bf*)bp, *(const v8bf*)(bp + 8));
        acc[s] = __builtin_amdgcn_wmma_f32_16x16x32_bf16(false, afrag, false, bfrag,
                                                         (short)0, acc[s], false, false);
      }
      __syncthreads();   // slab consumed
    }

    if (lane < 16) {
      const bf16_t* xg = a.XG + (size_t)t * B * GG;
#pragma unroll
      for (int u = 0; u < 2; ++u) {
        const int j = (w + 16 * u) * 16 + sub;
#pragma unroll
        for (int r = 0; r < 5; ++r) {
          if (r < B) {
            const bf16_t* xr = xg + (size_t)r * GG;
            float gi = acc[0 + u][r] + (float)xr[j];
            float gf = acc[2 + u][r] + (float)xr[512 + j];
            float gg = acc[4 + u][r] + (float)xr[1024 + j];
            float go = acc[6 + u][r] + (float)xr[1536 + j];
            float c  = sigm(gf) * cst[u][r] + sigm(gi) * tanhf(gg);
            cst[u][r] = c;
            float h  = sigm(go) * tanhf(c);
            H[r * CC + j] = (bf16_t)h;
            a.Hout[((size_t)t * B + r) * CC + j] = h;
          }
        }
      }
    }
    __syncthreads();
  }
}

// ---------------------------------------------------------------------------
// K3: attention scan. qe_t = P[t] + W2 @ out_t, W2 GEMV on the WMMA pipe.
// One block, 512 threads (16 waves); wave w owns n-tiles {2w, 2w+1}.
// ---------------------------------------------------------------------------
__global__ void __launch_bounds__(512) k_attn(const float* __restrict__ Hk,
                                              const float* __restrict__ Hv,
                                              const float* __restrict__ P,
                                              const bf16_t* __restrict__ W2,
                                              float* __restrict__ out) {
  __shared__ bf16_t abuf[16 * CC];   // A tile: row 0 = out_t, rows 1..15 zero
  __shared__ float  qe_l[CC];
  __shared__ float  red[CC];

  const int c    = threadIdx.x;
  const int lane = threadIdx.x & 31;
  const int w    = threadIdx.x >> 5;
  const int sub  = lane & 15;
  const int kh   = lane >> 4;

  for (int i = c; i < 16 * CC; i += 512) abuf[i] = (bf16_t)0.0f;

  float att[MM] = {1.f, 1.f, 1.f, 1.f, 1.f};
  float s[MM]   = {0.f, 0.f, 0.f, 0.f, 0.f};
  float pw = 1.0f;
  const float mom = 0.05f, om = 0.95f;

  const bf16_t* b0row = W2 + (size_t)((2 * w) * 16 + sub) * CC;
  const bf16_t* b1row = b0row + (size_t)16 * CC;
  __syncthreads();

  for (int t = 0; t < TT; ++t) {
    // out_t from register-resident att
    float oc = 0.f;
#pragma unroll
    for (int m = 0; m < MM; ++m) oc += att[m] * Hv[((size_t)t * MM + m) * CC + c];
    abuf[c] = (bf16_t)oc;             // A row 0
    __syncthreads();

    // W2 @ out_t on the matrix pipe: 2 n-tiles per wave, K=512
    v8f acc0 = {0.f, 0.f, 0.f, 0.f, 0.f, 0.f, 0.f, 0.f};
    v8f acc1 = {0.f, 0.f, 0.f, 0.f, 0.f, 0.f, 0.f, 0.f};
    for (int kc = 0; kc < 16; ++kc) {
      const bf16_t* ap = abuf + sub * CC + kc * 32 + kh * 16;
      v16bf a = cat8(*(const v8bf*)ap, *(const v8bf*)(ap + 8));
      const bf16_t* bp0 = b0row + kc * 32 + kh * 16;
      v16bf b0 = cat8(*(const v8bf*)bp0, *(const v8bf*)(bp0 + 8));
      acc0 = __builtin_amdgcn_wmma_f32_16x16x32_bf16(false, a, false, b0, (short)0, acc0, false, false);
      const bf16_t* bp1 = b1row + kc * 32 + kh * 16;
      v16bf b1 = cat8(*(const v8bf*)bp1, *(const v8bf*)(bp1 + 8));
      acc1 = __builtin_amdgcn_wmma_f32_16x16x32_bf16(false, a, false, b1, (short)0, acc1, false, false);
    }
    if (lane < 16) {                  // D row 0 lives in element 0 of lanes 0..15
      qe_l[(2 * w) * 16 + sub]     = acc0[0];
      qe_l[(2 * w + 1) * 16 + sub] = acc1[0];
    }
    __syncthreads();

    const float qe = P[(size_t)t * CC + c] + qe_l[c];

    pw *= om;
    const float inv = 1.0f / (1.0f - pw);
    float ab[MM];
    float part = 0.f;
#pragma unroll
    for (int m = 0; m < MM; ++m) {
      float av = qe * Hk[((size_t)t * MM + m) * CC + c];
      s[m] = om * s[m] + mom * av;
      ab[m] = s[m] * inv;
      part += ab[m] * ab[m];
    }
    red[c] = part;
    __syncthreads();
    for (int off = 256; off > 0; off >>= 1) {
      if (c < off) red[c] += red[c + off];
      __syncthreads();
    }
    const float nrm = sqrtf(red[0]) + 1e-12f;
#pragma unroll
    for (int m = 0; m < MM; ++m) att[m] = ab[m] / nrm;
    out[(size_t)t * CC + c] = oc;
    __syncthreads();
  }
}

// ---------------------------------------------------------------------------
// Host: workspace layout + launches
// ---------------------------------------------------------------------------
static constexpr size_t B_W   = (size_t)GG * CC * 2;            // bf16 weight matrix bytes
static constexpr size_t O_WHH_K = 0;
static constexpr size_t O_WHH_Q = O_WHH_K + B_W;
static constexpr size_t O_WHH_V = O_WHH_Q + B_W;
static constexpr size_t O_WIH_K = O_WHH_V + B_W;
static constexpr size_t O_WIH_Q = O_WIH_K + B_W;
static constexpr size_t O_WIH_V = O_WIH_Q + B_W;
static constexpr size_t O_BIAS_K = O_WIH_V + B_W;
static constexpr size_t O_BIAS_Q = O_BIAS_K + GG * 4;
static constexpr size_t O_BIAS_V = O_BIAS_Q + GG * 4;
static constexpr size_t O_XBF  = O_BIAS_V + GG * 4;
static constexpr size_t O_YBF  = O_XBF + (size_t)MM * TT * CC * 2;
static constexpr size_t O_XG_K = O_YBF + (size_t)TT * CC * 2;
static constexpr size_t O_XG_V = O_XG_K + (size_t)TT * MM * GG * 2;
static constexpr size_t O_XG_Q = O_XG_V + (size_t)TT * MM * GG * 2;
static constexpr size_t O_H_K  = O_XG_Q + (size_t)TT * GG * 2;
static constexpr size_t O_H_V  = O_H_K + (size_t)TT * MM * CC * 4;
static constexpr size_t O_H_Q  = O_H_V + (size_t)TT * MM * CC * 4;
static constexpr size_t O_W1   = O_H_Q + (size_t)TT * CC * 4;
static constexpr size_t O_W2   = O_W1 + (size_t)CC * CC * 2;
static constexpr size_t O_HQB  = O_W2 + (size_t)CC * CC * 2;
static constexpr size_t O_P    = O_HQB + (size_t)TT * CC * 2;
static constexpr size_t WS_NEED = O_P + (size_t)TT * CC * 4;

extern "C" void kernel_launch(void* const* d_in, const int* in_sizes, int n_in,
                              void* d_out, int out_size, void* d_ws, size_t ws_size,
                              hipStream_t stream) {
  (void)in_sizes; (void)n_in; (void)out_size;
  if (ws_size < WS_NEED) return;

  const float* x      = (const float*)d_in[0];
  const float* y      = (const float*)d_in[1];
  const float* Wk_ih  = (const float*)d_in[2];
  const float* Wk_hh  = (const float*)d_in[3];
  const float* Wk_bih = (const float*)d_in[4];
  const float* Wk_bhh = (const float*)d_in[5];
  const float* Wq_ih  = (const float*)d_in[6];
  const float* Wq_hh  = (const float*)d_in[7];
  const float* Wq_bih = (const float*)d_in[8];
  const float* Wq_bhh = (const float*)d_in[9];
  const float* Wv_ih  = (const float*)d_in[10];
  const float* Wv_hh  = (const float*)d_in[11];
  const float* Wv_bih = (const float*)d_in[12];
  const float* Wv_bhh = (const float*)d_in[13];
  const float* We_w   = (const float*)d_in[14];
  const float* We_b   = (const float*)d_in[15];

  char* ws = (char*)d_ws;
  bf16_t* whh_k = (bf16_t*)(ws + O_WHH_K);
  bf16_t* whh_q = (bf16_t*)(ws + O_WHH_Q);
  bf16_t* whh_v = (bf16_t*)(ws + O_WHH_V);
  bf16_t* wih_k = (bf16_t*)(ws + O_WIH_K);
  bf16_t* wih_q = (bf16_t*)(ws + O_WIH_Q);
  bf16_t* wih_v = (bf16_t*)(ws + O_WIH_V);
  float*  bias_k = (float*)(ws + O_BIAS_K);
  float*  bias_q = (float*)(ws + O_BIAS_Q);
  float*  bias_v = (float*)(ws + O_BIAS_V);
  bf16_t* xbf = (bf16_t*)(ws + O_XBF);
  bf16_t* ybf = (bf16_t*)(ws + O_YBF);
  bf16_t* xg_k = (bf16_t*)(ws + O_XG_K);
  bf16_t* xg_v = (bf16_t*)(ws + O_XG_V);
  bf16_t* xg_q = (bf16_t*)(ws + O_XG_Q);
  float*  h_k = (float*)(ws + O_H_K);
  float*  h_v = (float*)(ws + O_H_V);
  float*  h_q = (float*)(ws + O_H_Q);
  bf16_t* w1bf = (bf16_t*)(ws + O_W1);
  bf16_t* w2bf = (bf16_t*)(ws + O_W2);
  bf16_t* hq_bf = (bf16_t*)(ws + O_HQB);
  float*  p_buf = (float*)(ws + O_P);

  const int NW = GG * CC;
  dim3 cb(256);
  k_cvt_bf16<<<512, cb, 0, stream>>>(Wk_hh, whh_k, NW);
  k_cvt_bf16<<<512, cb, 0, stream>>>(Wq_hh, whh_q, NW);
  k_cvt_bf16<<<512, cb, 0, stream>>>(Wv_hh, whh_v, NW);
  k_cvt_bf16<<<512, cb, 0, stream>>>(Wk_ih, wih_k, NW);
  k_cvt_bf16<<<512, cb, 0, stream>>>(Wq_ih, wih_q, NW);
  k_cvt_bf16<<<512, cb, 0, stream>>>(Wv_ih, wih_v, NW);
  k_cvt_bf16<<<1024, cb, 0, stream>>>(x, xbf, MM * TT * CC);
  k_cvt_bf16<<<512, cb, 0, stream>>>(y, ybf, TT * CC);
  k_cvt_bf16_strided<<<256, cb, 0, stream>>>(We_w, w1bf, CC, CC, 2 * CC, 0);
  k_cvt_bf16_strided<<<256, cb, 0, stream>>>(We_w, w2bf, CC, CC, 2 * CC, CC);
  k_bias_sum<<<8, cb, 0, stream>>>(Wk_bih, Wk_bhh, bias_k, GG);
  k_bias_sum<<<8, cb, 0, stream>>>(Wq_bih, Wq_bhh, bias_q, GG);
  k_bias_sum<<<8, cb, 0, stream>>>(Wv_bih, Wv_bhh, bias_v, GG);

  // Input projections (pregates)
  k_input_proj<<<dim3(16, (MM * TT) / 16), 256, 0, stream>>>(xbf, wih_k, bias_k, xg_k, MM);
  k_input_proj<<<dim3(16, (MM * TT) / 16), 256, 0, stream>>>(xbf, wih_v, bias_v, xg_v, MM);
  k_input_proj<<<dim3(16, TT / 16),        256, 0, stream>>>(ybf, wih_q, bias_q, xg_q, 1);

  // Recurrent LSTMs: 3 persistent blocks, TDM-fed LDS double buffer (272KB)
  (void)hipFuncSetAttribute(reinterpret_cast<const void*>(&k_lstm),
                            hipFuncAttributeMaxDynamicSharedMemorySize, (int)DYN_LDS);
  LstmArgs ak{whh_k, xg_k, h_k, MM};
  LstmArgs av{whh_v, xg_v, h_v, MM};
  LstmArgs aq{whh_q, xg_q, h_q, 1};
  k_lstm<<<3, 512, DYN_LDS, stream>>>(ak, av, aq);

  // Precompute P[t] = Hq @ W1^T + Web (parallel WMMA GEMM)
  k_cvt_bf16<<<512, cb, 0, stream>>>(h_q, hq_bf, TT * CC);
  k_gemm_p<<<dim3(4, TT / 16), 256, 0, stream>>>(hq_bf, w1bf, We_b, p_buf);

  // Attention scan with WMMA GEMV for the recurrent half
  k_attn<<<1, 512, 0, stream>>>(h_k, h_v, p_buf, w2bf, (float*)d_out);
}
